// VGAEModel_65420941853303
// MI455X (gfx1250) — compile-verified, matrix operand order checked by reading, not run on hardware
//
#include <hip/hip_runtime.h>

// ---------------------------------------------------------------------------
// VGAE (3x GAT + reparam + sigmoid(z z^T)) for gfx1250 (MI455X, wave32, WMMA)
//
// v4: GEMM kernel templated on (K, Nn) so ksteps is compile-time -> the
// k-loop fully unrolls into straight-line code: no double-buffer register
// rotation (SSA renaming), every fragment load is base+imm24 offset, and the
// scheduler can overlap load clauses with WMMAs (partial loadcnt waits)
// instead of draining loadcnt to 0 across a loop back-edge.
// bf16 operands converted/packed once; adj written with NT stores
// (400 MB write-once stream vs 192 MB L2).
// ---------------------------------------------------------------------------

#define N_NODES   10000
#define E_EDGES   320000
#define IN_DIM    512
#define HID       256
#define ZDIM      64
#define HEADS     8
#define NEG_SLOPE 0.2f

typedef __attribute__((ext_vector_type(16))) __bf16        v16bf;
typedef __attribute__((ext_vector_type(8)))  float         v8f;
typedef __attribute__((ext_vector_type(4)))  unsigned int  u32x4;

// One WMMA operand fragment: 16 bf16 per lane.
union Frag { v16bf v; unsigned short u[16]; u32x4 q[2]; };

// Row-style fragment: elements p[0..7] and p[16..23] (ISA 16-bit A layout).
__device__ __forceinline__ void load_fragA(Frag& f, const unsigned short* p) {
  f.q[0] = *(const u32x4*)(p);
  f.q[1] = *(const u32x4*)(p + 16);
}
// Packed fragment: 16 contiguous elements.
__device__ __forceinline__ void load_fragP(Frag& f, const unsigned short* p) {
  f.q[0] = *(const u32x4*)(p);
  f.q[1] = *(const u32x4*)(p + 8);
}

__device__ __forceinline__ unsigned short f2bf(float f) {
  union { float f; unsigned u; } c; c.f = f;
  unsigned u = c.u + 0x7FFFu + ((c.u >> 16) & 1u);   // round-to-nearest-even
  return (unsigned short)(u >> 16);
}

__device__ __forceinline__ void atomicMaxF(float* addr, float val) {
  int* ia = (int*)addr;
  int old = __float_as_int(*addr);
  while (__int_as_float(old) < val) {
    int assumed = old;
    old = atomicCAS(ia, assumed, __float_as_int(val));
    if (old == assumed) break;
  }
}

// --------------------------- one-shot converters ---------------------------

__global__ void cvt_f32_bf16(const float* __restrict__ x,
                             unsigned short* __restrict__ y, int n) {
  int i = blockIdx.x * blockDim.x + threadIdx.x;
  if (i < n) y[i] = f2bf(x[i]);
}

// Pack f32 B[K,Nn] into bf16 WMMA B-fragments (512 contiguous bf16 each,
// lane-major, 16 per lane; lane (half,l16) covers column tn*16+l16 with
// K = kstep*32 + half*8 + {0..7} (j<8) / +16 more (j>=8)).
__global__ void pack_b_bf16(const float* __restrict__ B,
                            unsigned short* __restrict__ PB, int K, int Nn) {
  int idx = blockIdx.x * blockDim.x + threadIdx.x;
  if (idx >= K * Nn) return;
  int tilesN = Nn >> 4;
  int frag = idx >> 9;
  int r    = idx & 511;
  int lane = r >> 4, j = r & 15;
  int kstep = frag / tilesN, tn = frag % tilesN;
  int half = lane >> 4, l16 = lane & 15;
  int k   = kstep * 32 + half * 8 + ((j < 8) ? j : (16 + j - 8));
  int col = tn * 16 + l16;
  PB[idx] = f2bf(B[(size_t)k * Nn + col]);
}

// ---------------------------------------------------------------------------
// C[M,Nn] = A[M,K] @ B[K,Nn]; A bf16 row-major, B pre-packed fragments.
// Wave computes a 16x64 strip (4 N-tiles). K, Nn compile-time: fully
// unrolled straight-line body, all loads base+imm offset.
// ---------------------------------------------------------------------------
template <int K, int Nn>
__global__ void gemm_bf16_wmma(const unsigned short* __restrict__ A,
                               const unsigned short* __restrict__ PB,
                               float* __restrict__ C, int M) {
  constexpr int tilesN  = Nn >> 4;
  constexpr int nstrips = tilesN >> 2;
  constexpr int ksteps  = K >> 5;
  int wave = blockIdx.x * (blockDim.x >> 5) + (threadIdx.x >> 5);
  if (wave >= (M >> 4) * nstrips) return;          // wave-uniform
  int tm = wave / nstrips, strip = wave % nstrips;
  int lane = threadIdx.x & 31;
  int half = lane >> 4, l16 = lane & 15;

  const unsigned short* ap =
      A + (size_t)(tm * 16 + l16) * K + half * 8;
  const unsigned short* bp =
      PB + ((size_t)(strip * 4) << 9) + lane * 16;

  v8f acc[4] = {{}, {}, {}, {}};
#pragma unroll
  for (int ks = 0; ks < ksteps; ++ks) {
    Frag a;
    load_fragA(a, ap + ks * 32);
    if (ks + 2 < ksteps)
      __builtin_prefetch(ap + ks * 32 + 64, 0, 1); // global_prefetch_b8
    Frag b[4];
#pragma unroll
    for (int t = 0; t < 4; ++t)
      load_fragP(b[t], bp + ks * (tilesN * 512) + t * 512);
#pragma unroll
    for (int t = 0; t < 4; ++t)
      acc[t] = __builtin_amdgcn_wmma_f32_16x16x32_bf16(
          false, a.v, false, b[t].v, (short)0, acc[t], false, false);
  }

#pragma unroll
  for (int t = 0; t < 4; ++t) {
    int coln = (strip * 4 + t) * 16 + l16;
#pragma unroll
    for (int v = 0; v < 8; ++v)
      C[(size_t)(tm * 16 + v + half * 8) * Nn + coln] = acc[t][v];
  }
}

// ---------------------------------------------------------------------------
// adj[i,j] = sigmoid(z[i].z[j]); zb bf16 [N, ZDIM] row-major.
// Wave computes a 16x80 strip (5 tiles, 625 = 5*125); straight-line after
// unroll (constant trip count), NT stores for the adjacency stream.
// ---------------------------------------------------------------------------
#define ZTILES 5
__global__ void zzt_sigmoid_wmma(const unsigned short* __restrict__ zb,
                                 float* __restrict__ adj) {
  const int tiles   = N_NODES / 16;                // 625
  const int nstrips = tiles / ZTILES;              // 125
  int wave = blockIdx.x * (blockDim.x >> 5) + (threadIdx.x >> 5);
  if (wave >= tiles * nstrips) return;             // wave-uniform
  int tm = wave / nstrips, strip = wave % nstrips;
  int lane = threadIdx.x & 31;
  int half = lane >> 4, l16 = lane & 15;

  const unsigned short* ab =
      zb + (size_t)(tm * 16 + l16) * ZDIM + half * 8;
  Frag a0, a1;
  load_fragA(a0, ab);
  load_fragA(a1, ab + 32);

  const unsigned short* bb =
      zb + (size_t)(strip * ZTILES * 16 + l16) * ZDIM + half * 8;

  v8f acc[ZTILES] = {};
#pragma unroll
  for (int t = 0; t < ZTILES; ++t) {
    Frag b0, b1;
    const unsigned short* bn = bb + (size_t)t * 16 * ZDIM;
    load_fragA(b0, bn);
    load_fragA(b1, bn + 32);
    acc[t] = __builtin_amdgcn_wmma_f32_16x16x32_bf16(
        false, a0.v, false, b0.v, (short)0, acc[t], false, false);
    acc[t] = __builtin_amdgcn_wmma_f32_16x16x32_bf16(
        false, a1.v, false, b1.v, (short)0, acc[t], false, false);
  }

#pragma unroll
  for (int t = 0; t < ZTILES; ++t) {
    int coln = (strip * ZTILES + t) * 16 + l16;
#pragma unroll
    for (int v = 0; v < 8; ++v) {
      float x = acc[t][v];
      float sg = 1.f / (1.f + __expf(-x));
      __builtin_nontemporal_store(
          sg, &adj[(size_t)(tm * 16 + v + half * 8) * N_NODES + coln]);
    }
  }
}

// --------------------------- elementwise / scatter -------------------------

__global__ void fill_f32(float* __restrict__ p, float v, int n) {
  int i = blockIdx.x * blockDim.x + threadIdx.x;
  if (i < n) p[i] = v;
}

__global__ void attn_scores(const float* __restrict__ h,
                            const float* __restrict__ al,
                            const float* __restrict__ ar,
                            float* __restrict__ el, float* __restrict__ er,
                            int D) {
  int i = blockIdx.x * blockDim.x + threadIdx.x;
  if (i >= N_NODES * HEADS) return;
  int node = i / HEADS, head = i % HEADS;
  const float* hp  = h  + (size_t)node * HEADS * D + head * D;
  const float* alp = al + head * D;
  const float* arp = ar + head * D;
  float sl = 0.f, sr = 0.f;
  for (int d = 0; d < D; ++d) { sl += hp[d] * alp[d]; sr += hp[d] * arp[d]; }
  el[i] = sl; er[i] = sr;
}

__global__ void init_ms(float* __restrict__ m, float* __restrict__ s) {
  int i = blockIdx.x * blockDim.x + threadIdx.x;
  if (i < N_NODES * HEADS) { m[i] = -3.0e38f; s[i] = 0.f; }
}

__global__ void edge_score(const int* __restrict__ src, const int* __restrict__ dst,
                           const float* __restrict__ el, const float* __restrict__ er,
                           float* __restrict__ ev, float* __restrict__ m) {
  int i = blockIdx.x * blockDim.x + threadIdx.x;
  if (i >= E_EDGES * HEADS) return;
  int e = i / HEADS, h = i % HEADS;
  float v = el[src[e] * HEADS + h] + er[dst[e] * HEADS + h];
  v = v > 0.f ? v : NEG_SLOPE * v;
  ev[i] = v;
  atomicMaxF(&m[dst[e] * HEADS + h], v);
}

__global__ void edge_expsum(const int* __restrict__ dst,
                            float* __restrict__ ev,
                            const float* __restrict__ m,
                            float* __restrict__ s) {
  int i = blockIdx.x * blockDim.x + threadIdx.x;
  if (i >= E_EDGES * HEADS) return;
  int e = i / HEADS, h = i % HEADS;
  int didx = dst[e] * HEADS + h;
  float v = __expf(ev[i] - m[didx]);
  ev[i] = v;
  atomicAdd(&s[didx], v);
}

__global__ void edge_aggregate(const int* __restrict__ src, const int* __restrict__ dst,
                               const float* __restrict__ hfeat,
                               const float* __restrict__ ev,
                               const float* __restrict__ s,
                               float* __restrict__ agg, int D) {
  int i = blockIdx.x * blockDim.x + threadIdx.x;
  if (i >= E_EDGES * HEADS * D) return;
  int d  = i % D;
  int eh = i / D;
  int e  = eh / HEADS, h = eh % HEADS;
  int sN = src[e], dN = dst[e];
  float alpha = ev[eh] / s[dN * HEADS + h];
  float msg = hfeat[(size_t)sN * HEADS * D + h * D + d] * alpha;
  atomicAdd(&agg[(size_t)dN * HEADS * D + h * D + d], msg);
}

__global__ void bias_act(float* __restrict__ x, const float* __restrict__ b,
                         int n, int F, int do_relu) {
  int i = blockIdx.x * blockDim.x + threadIdx.x;
  if (i >= n * F) return;
  float v = x[i] + b[i % F];
  x[i] = do_relu ? fmaxf(v, 0.f) : v;
}

__global__ void reparam(const float* __restrict__ mean,
                        const float* __restrict__ logstd,
                        const float* __restrict__ noise,
                        float* __restrict__ z, unsigned short* __restrict__ zb) {
  int i = blockIdx.x * blockDim.x + threadIdx.x;
  if (i >= N_NODES * ZDIM) return;
  float v = mean[i] + noise[i] * __expf(logstd[i]);
  z[i] = v;
  zb[i] = f2bf(v);
}

// ---------------------------------------------------------------------------

static inline int ceil_div(int a, int b) { return (a + b - 1) / b; }

extern "C" void kernel_launch(void* const* d_in, const int* in_sizes, int n_in,
                              void* d_out, int out_size, void* d_ws, size_t ws_size,
                              hipStream_t stream) {
  (void)in_sizes; (void)n_in; (void)out_size; (void)ws_size;

  const float* features = (const float*)d_in[0];
  const int*   src      = (const int*)  d_in[1];
  const int*   dst      = (const int*)  d_in[2];
  const float* noise    = (const float*)d_in[3];
  const float* W0  = (const float*)d_in[4];
  const float* al0 = (const float*)d_in[5];
  const float* ar0 = (const float*)d_in[6];
  const float* b0  = (const float*)d_in[7];
  const float* W1  = (const float*)d_in[8];
  const float* al1 = (const float*)d_in[9];
  const float* ar1 = (const float*)d_in[10];
  const float* b1  = (const float*)d_in[11];
  const float* W2  = (const float*)d_in[12];
  const float* al2 = (const float*)d_in[13];
  const float* ar2 = (const float*)d_in[14];
  const float* b2  = (const float*)d_in[15];

  // ---- workspace layout: f32 region first (16B-aligned total), bf16 after
  float* ws   = (float*)d_ws;
  float* hbuf = ws;                                   // N*HID
  float* hact = hbuf + (size_t)N_NODES * HID;         // N*HID
  float* g1   = hact + (size_t)N_NODES * HID;         // N*Z
  float* a1   = g1   + (size_t)N_NODES * ZDIM;
  float* g2   = a1   + (size_t)N_NODES * ZDIM;
  float* a2   = g2   + (size_t)N_NODES * ZDIM;
  float* el   = a2   + (size_t)N_NODES * ZDIM;        // N*H
  float* er   = el + (size_t)N_NODES * HEADS;
  float* m    = er + (size_t)N_NODES * HEADS;
  float* s    = m  + (size_t)N_NODES * HEADS;
  float* ev   = s  + (size_t)N_NODES * HEADS;         // E*H
  unsigned short* xb    = (unsigned short*)(ev + (size_t)E_EDGES * HEADS);
  unsigned short* hactb = xb    + (size_t)N_NODES * IN_DIM;   // N*HID
  unsigned short* W0p   = hactb + (size_t)N_NODES * HID;      // 512*256
  unsigned short* W1p   = W0p   + (size_t)IN_DIM * HID;       // 256*64
  unsigned short* W2p   = W1p   + (size_t)HID * ZDIM;
  unsigned short* zb    = W2p   + (size_t)HID * ZDIM;         // N*64

  const int TB = 256;   // elementwise block
  const int WB = 128;   // 4 waves/block for WMMA kernels

  // one-shot operand preparation (bf16 + fragment packing)
  pack_b_bf16<<<ceil_div(IN_DIM * HID, TB), TB, 0, stream>>>(W0, W0p, IN_DIM, HID);
  pack_b_bf16<<<ceil_div(HID * ZDIM, TB), TB, 0, stream>>>(W1, W1p, HID, ZDIM);
  pack_b_bf16<<<ceil_div(HID * ZDIM, TB), TB, 0, stream>>>(W2, W2p, HID, ZDIM);
  cvt_f32_bf16<<<ceil_div(N_NODES * IN_DIM, TB), TB, 0, stream>>>(
      features, xb, N_NODES * IN_DIM);

  // common GAT tail (attention + edge softmax + scatter + bias/act)
  auto gat_tail = [&](const float* hout, const float* al, const float* ar,
                      const float* bias, int D, float* agg, int do_relu) {
    int Nn = HEADS * D;
    attn_scores<<<ceil_div(N_NODES * HEADS, TB), TB, 0, stream>>>(
        hout, al, ar, el, er, D);
    init_ms<<<ceil_div(N_NODES * HEADS, TB), TB, 0, stream>>>(m, s);
    edge_score<<<ceil_div(E_EDGES * HEADS, TB), TB, 0, stream>>>(
        src, dst, el, er, ev, m);
    edge_expsum<<<ceil_div(E_EDGES * HEADS, TB), TB, 0, stream>>>(dst, ev, m, s);
    fill_f32<<<ceil_div(N_NODES * Nn, TB), TB, 0, stream>>>(agg, 0.f, N_NODES * Nn);
    edge_aggregate<<<ceil_div(E_EDGES * HEADS * D, TB), TB, 0, stream>>>(
        src, dst, hout, ev, s, agg, D);
    bias_act<<<ceil_div(N_NODES * Nn, TB), TB, 0, stream>>>(
        agg, bias, N_NODES, Nn, do_relu);
  };

  // layer 0: features -> hact (relu)
  {
    int nwaves = (N_NODES / 16) * ((HID / 16) / 4);
    gemm_bf16_wmma<IN_DIM, HID><<<ceil_div(nwaves, WB / 32), WB, 0, stream>>>(
        xb, W0p, hbuf, N_NODES);
    gat_tail(hbuf, al0, ar0, b0, HID / HEADS, hact, 1);
  }
  // bf16 copy of hact for layers 1/2
  cvt_f32_bf16<<<ceil_div(N_NODES * HID, TB), TB, 0, stream>>>(
      hact, hactb, N_NODES * HID);
  // layer 1 (mean) and layer 2 (log_std)
  {
    int nwaves = (N_NODES / 16) * ((ZDIM / 16) / 4);
    gemm_bf16_wmma<HID, ZDIM><<<ceil_div(nwaves, WB / 32), WB, 0, stream>>>(
        hactb, W1p, g1, N_NODES);
    gat_tail(g1, al1, ar1, b1, ZDIM / HEADS, a1, 0);
    gemm_bf16_wmma<HID, ZDIM><<<ceil_div(nwaves, WB / 32), WB, 0, stream>>>(
        hactb, W2p, g2, N_NODES);
    gat_tail(g2, al2, ar2, b2, ZDIM / HEADS, a2, 0);
  }

  // z = mean + noise*exp(log_std) -> d_out[0 : N*Z] (+ bf16 copy)
  float* z_out = (float*)d_out;
  reparam<<<ceil_div(N_NODES * ZDIM, TB), TB, 0, stream>>>(a1, a2, noise, z_out, zb);

  // adj = sigmoid(z @ z^T) -> d_out[N*Z : ]
  {
    float* adj = z_out + (size_t)N_NODES * ZDIM;
    int nwaves = (N_NODES / 16) * ((N_NODES / 16) / ZTILES);
    zzt_sigmoid_wmma<<<ceil_div(nwaves, WB / 32), WB, 0, stream>>>(zb, adj);
  }
}
